// AugustBlock_42880953483913
// MI455X (gfx1250) — matrix operand order repaired and checked by reading, not compiled, not run on hardware
//
#include <hip/hip_runtime.h>
#include <hip/hip_bf16.h>
#include <math.h>

// ---------------------------------------------------------------------------
// SwinV2-3D block for MI455X (gfx1250, wave32, WMMA 16x16x32 f16 -> f32 acc)
// ---------------------------------------------------------------------------

typedef __attribute__((ext_vector_type(16))) _Float16 v16h;
typedef __attribute__((ext_vector_type(8)))  float    v8f;

#define DEV static __device__ __forceinline__

// ---- problem constants ----
constexpr int RESV = 32;          // cubic volume side
constexpr int WS   = 8;           // window size
constexpr int SH   = 4;           // shift
constexpr int NTOK = WS*WS*WS;    // 512 tokens / window
constexpr int NW   = 64;          // (32/8)^3 windows
constexpr int CDIM = 128;
constexpr int NH   = 4;
constexpr int HD   = 32;
constexpr int TOKS = NW*NTOK;     // 32768 tokens
constexpr int CPB_L = 2*WS-1;     // 15
constexpr int CPB_E = CPB_L*CPB_L*CPB_L; // 3375
constexpr int HID  = 512;         // cpb hidden & mlp hidden

// ---------------------------------------------------------------------------
// WMMA helpers
// ---------------------------------------------------------------------------
DEV v8f wmma_f16(v16h a, v16h b, v8f c) {
  // D = A(16x32) * B(32x16) + C, f32 accumulate
  return __builtin_amdgcn_wmma_f32_16x16x32_f16(
      /*neg_a=*/false, a, /*neg_b=*/false, b,
      /*c_mod=*/(short)0, c, /*reuse_a=*/false, /*reuse_b=*/false);
}

// A fragment: src row-major (16 rows x K), lane(hf=lane>>4, r=lane&15)
// elem j -> K = hf*16+j   (row r)
DEV v16h load_frag_a(const _Float16* base, int ld) {
  const int lane = threadIdx.x & 31, r = lane & 15, hf = lane >> 4;
  const _Float16* p = base + (size_t)r * ld + hf * 16;
  v16h a;
#pragma unroll
  for (int j = 0; j < 8; ++j) a[j] = p[j];
#pragma unroll
  for (int j = 0; j < 8; ++j) a[8 + j] = p[8 + j];
  return a;
}

// B fragment: source stored TRANSPOSED row-major (N x K).
// lane(hf, n=r): elem j -> B[K=hf*16+j][n] = srcT[n][hf*16+j]
DEV v16h load_frag_b(const _Float16* baseT, int ld) {
  const int lane = threadIdx.x & 31, r = lane & 15, hf = lane >> 4;
  const _Float16* p = baseT + (size_t)r * ld + hf * 16;
  v16h b;
#pragma unroll
  for (int j = 0; j < 16; ++j) b[j] = p[j];
  return b;
}

// Register-blocked GEMM: one wave computes a 16 x 64 output slab
// (4 consecutive N-tiles). Per K-step: issue ALL fragment loads first,
// then a sched_barrier pins them above the 4 WMMAs so the waiter emits
// descending s_wait_loadcnt (loads overlap matrix math). 4 accumulators
// (32 VGPRs) + 4 B frags (32) + A (8) stays within the VGPR budget --
// the 8-wide variant spilled accumulators to scratch.
template <int KDIM>
DEV void gemm_slab4(const _Float16* __restrict__ Ab,   // 16 x KDIM rows
                    const _Float16* __restrict__ Bb,   // (4*16) x KDIM (N,K)
                    v8f c[4]) {
  for (int k0 = 0; k0 < KDIM; k0 += 32) {
    v16h a = load_frag_a(Ab + k0, KDIM);
    v16h bf[4];
#pragma unroll
    for (int t = 0; t < 4; ++t)
      bf[t] = load_frag_b(Bb + (size_t)t * 16 * KDIM + k0, KDIM);
    __builtin_amdgcn_sched_barrier(0);   // keep the load burst above the math
#pragma unroll
    for (int t = 0; t < 4; ++t)
      c[t] = wmma_f16(a, bf[t], c[t]);
  }
}

// ---------------------------------------------------------------------------
// 0) f32 -> f16 weight convert
// ---------------------------------------------------------------------------
__global__ void k_f32_to_f16(const float* __restrict__ src,
                             _Float16* __restrict__ dst, int n) {
  int i = blockIdx.x * 256 + threadIdx.x;
  if (i < n) dst[i] = (_Float16)src[i];
}

// ---------------------------------------------------------------------------
// 1) CPB MLP: bt[e][h] = (relu(ctab[e] @ w1^T + b1) @ w2^T)   (3375 x 4)
// ---------------------------------------------------------------------------
__global__ void k_cpb(const float* __restrict__ w1, const float* __restrict__ b1,
                      const float* __restrict__ w2, float* __restrict__ bt) {
  __shared__ float red[HID];
  const int e = blockIdx.x;       // 0..3374
  const int j = threadIdx.x;      // 0..511
  const int i0 = e / (CPB_L*CPB_L), i1 = (e / CPB_L) % CPB_L, i2 = e % CPB_L;
  float c[3];
  const int ii[3] = {i0, i1, i2};
#pragma unroll
  for (int d = 0; d < 3; ++d) {
    float rr = (float)(ii[d] - (WS - 1));
    float t = rr / (float)(WS - 1) * 8.0f;
    float s = (t > 0.f) ? 1.f : ((t < 0.f) ? -1.f : 0.f);
    c[d] = s * log2f(fabsf(t) + 1.0f) / 3.0f;   // log2(8)=3
  }
  float hid = w1[j*3+0]*c[0] + w1[j*3+1]*c[1] + w1[j*3+2]*c[2] + b1[j];
  hid = fmaxf(hid, 0.0f);
  for (int h = 0; h < NH; ++h) {
    red[j] = hid * w2[h * HID + j];
    __syncthreads();
    for (int s = HID / 2; s > 0; s >>= 1) {
      if (j < s) red[j] += red[j + s];
      __syncthreads();
    }
    if (j == 0) bt[e * NH + h] = red[0];
    __syncthreads();
  }
}

// ---------------------------------------------------------------------------
// 2) Expand bias: BIAS[h][n][m] = 16*sigmoid(bt[relidx(n,m)][h])
// ---------------------------------------------------------------------------
__global__ void k_bias(const float* __restrict__ bt, float* __restrict__ BIAS) {
  int idx = blockIdx.x * 256 + threadIdx.x;     // 0..262143
  int n = idx >> 9, m = idx & 511;
  int zn = n >> 6, yn = (n >> 3) & 7, xn = n & 7;
  int zm = m >> 6, ym = (m >> 3) & 7, xm = m & 7;
  int rz = zn - zm + (WS - 1), ry = yn - ym + (WS - 1), rx = xn - xm + (WS - 1);
  int ridx = (rz * CPB_L + ry) * CPB_L + rx;
#pragma unroll
  for (int h = 0; h < NH; ++h) {
    float v = bt[ridx * NH + h];
    BIAS[((size_t)(h * NTOK + n)) * NTOK + m] = 16.0f / (1.0f + expf(-v));
  }
}

// ---------------------------------------------------------------------------
// 3) cyclic shift + window partition, f32 -> f16  (8 channels / thread)
// ---------------------------------------------------------------------------
__global__ void k_gather(const float* __restrict__ x, _Float16* __restrict__ xw) {
  int e = blockIdx.x * 256 + threadIdx.x;       // TOKS*16 = 524288
  int c8 = (e & 15) * 8;
  int t  = e >> 4;
  int w = t >> 9, n = t & 511;
  int wd = w >> 4, wh = (w >> 2) & 3, ww = w & 3;
  int z = n >> 6, y = (n >> 3) & 7, xx = n & 7;
  int gz = (wd * WS + z + SH) & 31, gy = (wh * WS + y + SH) & 31,
      gx = (ww * WS + xx + SH) & 31;
  size_t g = ((size_t)gz * RESV + gy) * RESV + gx;
  const float* src = x + g * CDIM + c8;
  _Float16* dst = xw + (size_t)t * CDIM + c8;
#pragma unroll
  for (int j = 0; j < 8; ++j) dst[j] = (_Float16)src[j];
}

// ---------------------------------------------------------------------------
// 4) QKV GEMM: (512x128)@(128x384); each wave -> 16x64 slab.
//    write Q,K (n,d) and V^T (d,n)
// ---------------------------------------------------------------------------
__global__ void k_qkv(const _Float16* __restrict__ XW, const _Float16* __restrict__ Wt,
                      const float* __restrict__ bias,
                      _Float16* __restrict__ Q, _Float16* __restrict__ K,
                      _Float16* __restrict__ Vt) {
  const int tid = threadIdx.x, wave = tid >> 5, lane = tid & 31;
  const int r = lane & 15, hf = lane >> 4;
  int wid = blockIdx.x * 8 + wave;              // NW*32*6 = 12288 waves
  int ng = wid % 6; wid /= 6;                   // N-group (4 tiles of the 24)
  int mt = wid % 32; int w = wid / 32;
  const _Float16* Ab = XW + ((size_t)w * NTOK + mt * 16) * CDIM;
  const _Float16* Bb = Wt + (size_t)ng * 4 * 16 * CDIM;
  v8f c[4];
#pragma unroll
  for (int t = 0; t < 4; ++t) c[t] = (v8f){};
  gemm_slab4<CDIM>(Ab, Bb, c);
  size_t wbase = (size_t)w * NH;
#pragma unroll
  for (int t = 0; t < 4; ++t) {
    int nt = ng * 4 + t;
    int o = nt * 16 + r;
    float bv = bias[o];
    int which = o >> 7, h = (o >> 5) & 3, d = o & 31;
    size_t whb = wbase + h;
#pragma unroll
    for (int i = 0; i < 8; ++i) {
      int n = mt * 16 + hf * 8 + i;
      float val = c[t][i] + bv;
      if (which == 0)      Q[(whb * NTOK + n) * HD + d] = (_Float16)val;
      else if (which == 1) K[(whb * NTOK + n) * HD + d] = (_Float16)val;
      else                 Vt[(whb * HD + d) * NTOK + n] = (_Float16)val;
    }
  }
}

// ---------------------------------------------------------------------------
// 5) cosine-normalize rows of Q,K; fold per-head logit scale into Q
// ---------------------------------------------------------------------------
__global__ void k_norm(_Float16* __restrict__ Q, _Float16* __restrict__ K,
                       const float* __restrict__ ls) {
  int idx = blockIdx.x * 256 + threadIdx.x;     // NW*NH*NTOK rows
  int h = (idx >> 9) & 3;
  float scale = expf(fminf(ls[h], logf(100.0f)));
  _Float16* q = Q + (size_t)idx * HD;
  _Float16* k = K + (size_t)idx * HD;
  float sq = 0.f, sk = 0.f;
#pragma unroll
  for (int d = 0; d < HD; ++d) { float a=(float)q[d], b=(float)k[d]; sq+=a*a; sk+=b*b; }
  float iq = scale / (sqrtf(sq) + 1e-12f);
  float ik = 1.0f / (sqrtf(sk) + 1e-12f);
#pragma unroll
  for (int d = 0; d < HD; ++d) {
    q[d] = (_Float16)((float)q[d] * iq);
    k[d] = (_Float16)((float)k[d] * ik);
  }
}

// ---------------------------------------------------------------------------
// 6) attention per (window, head, 16-row block): QK^T -> +bias+mask ->
//    softmax -> P(f16) -> P@V    (all matmuls via WMMA)
// ---------------------------------------------------------------------------
DEV int region_of(int tok, int wd, int wh, int ww) {
  int z = tok >> 6, y = (tok >> 3) & 7, x = tok & 7;
  int rz = (wd == 3) ? ((z < SH) ? 1 : 2) : 0;
  int ry = (wh == 3) ? ((y < SH) ? 1 : 2) : 0;
  int rx = (ww == 3) ? ((x < SH) ? 1 : 2) : 0;
  return rz * 9 + ry * 3 + rx;
}

__global__ void k_attn(const _Float16* __restrict__ Q, const _Float16* __restrict__ K,
                       const _Float16* __restrict__ Vt, const float* __restrict__ BIAS,
                       _Float16* __restrict__ AO) {
  __shared__ float    S[16][NTOK];       // 32 KB
  __shared__ _Float16 P[16][NTOK];       // 16 KB
  __shared__ float    red[16][8];        // softmax reductions
  __shared__ float    Opart[2][16][16];  // AV split-K partials
  const int nb = blockIdx.x, h = blockIdx.y, w = blockIdx.z;
  const int tid = threadIdx.x, wave = tid >> 5, lane = tid & 31;
  const int r = lane & 15, hf = lane >> 4;
  const int n0 = nb * 16;
  const int wd = w >> 4, wh = (w >> 2) & 3, ww = w & 3;
  const size_t whb = (size_t)(w * NH + h);
  const _Float16* Qb = Q + whb * NTOK * HD;
  const _Float16* Kb = K + whb * NTOK * HD;
  const _Float16* Vb = Vt + whb * HD * NTOK;

  // ---- scores: 32 m-tiles over 4 waves, one wmma each (K = d = 32) ----
  // double-buffer the K-side fragment so the next load overlaps the wmma
  v16h aq = load_frag_a(Qb + (size_t)n0 * HD, HD);
  v16h bk = load_frag_b(Kb + (size_t)wave * 16 * HD, HD);
  for (int mt = wave; mt < 32; mt += 4) {
    v16h bk_next;
    if (mt + 4 < 32)
      bk_next = load_frag_b(Kb + (size_t)(mt + 4) * 16 * HD, HD);
    v8f c = {};
    c = wmma_f16(aq, bk, c);
    int m = mt * 16 + r;
    int regm = region_of(m, wd, wh, ww);
#pragma unroll
    for (int i = 0; i < 8; ++i) {
      int nrow = n0 + hf * 8 + i;
      int regn = region_of(nrow, wd, wh, ww);
      float bias = BIAS[((size_t)(h * NTOK + nrow)) * NTOK + m];
      S[hf * 8 + i][m] = c[i] + bias + ((regn == regm) ? 0.0f : -100.0f);
    }
    bk = bk_next;
  }
  __syncthreads();

  // ---- softmax over m: 8 threads per row ----
  {
    const int row = tid >> 3, sub = tid & 7;
    float mx = -1e30f;
    for (int j = sub; j < NTOK; j += 8) mx = fmaxf(mx, S[row][j]);
    red[row][sub] = mx;
    __syncthreads();
    if (sub == 0) {
      float m0 = red[row][0];
#pragma unroll
      for (int t = 1; t < 8; ++t) m0 = fmaxf(m0, red[row][t]);
      red[row][0] = m0;
    }
    __syncthreads();
    const float rowmax = red[row][0];
    __syncthreads();
    float sum = 0.f;
    for (int j = sub; j < NTOK; j += 8) {
      float e = expf(S[row][j] - rowmax);
      S[row][j] = e;
      sum += e;
    }
    red[row][sub] = sum;
    __syncthreads();
    if (sub == 0) {
      float s0 = 0.f;
#pragma unroll
      for (int t = 0; t < 8; ++t) s0 += red[row][t];
      red[row][0] = s0;
    }
    __syncthreads();
    const float inv = 1.0f / red[row][0];
    for (int j = sub; j < NTOK; j += 8)
      P[row][j] = (_Float16)(S[row][j] * inv);
  }
  __syncthreads();

  // ---- P @ V : 2 d-tiles x 2 K-halves across the 4 waves ----
  {
    const int dt = wave & 1;        // d-tile
    const int kh = wave >> 1;       // K half (m in [kh*256, kh*256+256))
    const int d0 = dt * 16;
    v8f c = {};
#pragma unroll
    for (int kk = 0; kk < 8; ++kk) {
      int k0 = kh * 256 + kk * 32;
      v16h pa = load_frag_a(&P[0][0] + k0, NTOK);
      v16h vb = load_frag_b(Vb + (size_t)d0 * NTOK + k0, NTOK);
      c = wmma_f16(pa, vb, c);
    }
    if (kh == 1) {
#pragma unroll
      for (int i = 0; i < 8; ++i) Opart[dt][hf * 8 + i][r] = c[i];
    }
    __syncthreads();
    if (kh == 0) {
#pragma unroll
      for (int i = 0; i < 8; ++i) {
        int ntok = n0 + hf * 8 + i;
        float v = c[i] + Opart[dt][hf * 8 + i][r];
        AO[((size_t)w * NTOK + ntok) * CDIM + h * HD + d0 + r] = (_Float16)v;
      }
    }
  }
}

// ---------------------------------------------------------------------------
// 7) slab GEMMs: proj (N=128), fc1 (N=512, gelu), fc2 (N=128, K=512)
// ---------------------------------------------------------------------------
__global__ void k_proj(const _Float16* __restrict__ A, const _Float16* __restrict__ Wt,
                       const float* __restrict__ bias, float* __restrict__ O) {
  const int tid = threadIdx.x, wave = tid >> 5, lane = tid & 31;
  const int r = lane & 15, hf = lane >> 4;
  int wid = blockIdx.x * 8 + wave;              // 4096 waves
  int ng = wid & 1, mt = wid >> 1;              // 2 N-groups of 4 tiles
  const _Float16* Ab = A + (size_t)mt * 16 * CDIM;
  const _Float16* Bb = Wt + (size_t)ng * 4 * 16 * CDIM;
  v8f c[4];
#pragma unroll
  for (int t = 0; t < 4; ++t) c[t] = (v8f){};
  gemm_slab4<CDIM>(Ab, Bb, c);
#pragma unroll
  for (int t = 0; t < 4; ++t) {
    int o = (ng * 4 + t) * 16 + r;
    float bv = bias[o];
#pragma unroll
    for (int i = 0; i < 8; ++i)
      O[((size_t)mt * 16 + hf * 8 + i) * CDIM + o] = c[t][i] + bv;
  }
}

__global__ void k_fc1(const _Float16* __restrict__ A, const _Float16* __restrict__ Wt,
                      const float* __restrict__ bias, _Float16* __restrict__ H) {
  const int tid = threadIdx.x, wave = tid >> 5, lane = tid & 31;
  const int r = lane & 15, hf = lane >> 4;
  int wid = blockIdx.x * 8 + wave;              // 16384 waves
  int ng = wid & 7, mt = wid >> 3;              // 8 N-groups of 4 tiles
  const _Float16* Ab = A + (size_t)mt * 16 * CDIM;
  const _Float16* Bb = Wt + (size_t)ng * 4 * 16 * CDIM;
  v8f c[4];
#pragma unroll
  for (int t = 0; t < 4; ++t) c[t] = (v8f){};
  gemm_slab4<CDIM>(Ab, Bb, c);
#pragma unroll
  for (int t = 0; t < 4; ++t) {
    int o = (ng * 4 + t) * 16 + r;
    float bv = bias[o];
#pragma unroll
    for (int i = 0; i < 8; ++i) {
      float v = c[t][i] + bv;
      // gelu (tanh approximation, jax default)
      float g = 0.5f * v * (1.0f + tanhf(0.7978845608f * (v + 0.044715f * v * v * v)));
      H[((size_t)mt * 16 + hf * 8 + i) * HID + o] = (_Float16)g;
    }
  }
}

__global__ void k_fc2(const _Float16* __restrict__ Hm, const _Float16* __restrict__ Wt,
                      const float* __restrict__ bias, float* __restrict__ O) {
  const int tid = threadIdx.x, wave = tid >> 5, lane = tid & 31;
  const int r = lane & 15, hf = lane >> 4;
  int wid = blockIdx.x * 8 + wave;              // 4096 waves
  int ng = wid & 1, mt = wid >> 1;              // 2 N-groups of 4 tiles
  const _Float16* Ab = Hm + (size_t)mt * 16 * HID;
  const _Float16* Bb = Wt + (size_t)ng * 4 * 16 * HID;
  v8f c[4];
#pragma unroll
  for (int t = 0; t < 4; ++t) c[t] = (v8f){};
  gemm_slab4<HID>(Ab, Bb, c);
#pragma unroll
  for (int t = 0; t < 4; ++t) {
    int o = (ng * 4 + t) * 16 + r;
    float bv = bias[o];
#pragma unroll
    for (int i = 0; i < 8; ++i)
      O[((size_t)mt * 16 + hf * 8 + i) * CDIM + o] = c[t][i] + bv;
  }
}

// ---------------------------------------------------------------------------
// 8) window-reverse + roll-back + residual + LayerNorm -> X1 (f32 & f16)
// ---------------------------------------------------------------------------
__global__ void k_ln1(const float* __restrict__ x, const float* __restrict__ PO,
                      const float* __restrict__ nw, const float* __restrict__ nb,
                      float* __restrict__ X1, _Float16* __restrict__ X116) {
  int t = blockIdx.x * 256 + threadIdx.x;       // window-order token
  int w = t >> 9, n = t & 511;
  int wd = w >> 4, wh = (w >> 2) & 3, ww = w & 3;
  int z = n >> 6, y = (n >> 3) & 7, xx = n & 7;
  int gz = (wd * WS + z + SH) & 31, gy = (wh * WS + y + SH) & 31,
      gx = (ww * WS + xx + SH) & 31;
  size_t g = ((size_t)gz * RESV + gy) * RESV + gx;
  const float* po = PO + (size_t)t * CDIM;
  float mu = 0.f;
#pragma unroll 4
  for (int c = 0; c < CDIM; ++c) mu += po[c];
  mu *= (1.0f / CDIM);
  float var = 0.f;
#pragma unroll 4
  for (int c = 0; c < CDIM; ++c) { float d = po[c] - mu; var += d * d; }
  float rstd = rsqrtf(var * (1.0f / CDIM) + 1e-5f);
#pragma unroll 4
  for (int c = 0; c < CDIM; ++c) {
    float v = x[g * CDIM + c] + (po[c] - mu) * rstd * nw[c] + nb[c];
    X1[g * CDIM + c] = v;
    X116[g * CDIM + c] = (_Float16)v;
  }
}

// ---------------------------------------------------------------------------
// 9) final residual + LayerNorm -> out
// ---------------------------------------------------------------------------
__global__ void k_ln2(const float* __restrict__ X1, const float* __restrict__ MO,
                      const float* __restrict__ nw, const float* __restrict__ nb,
                      float* __restrict__ out) {
  int t = blockIdx.x * 256 + threadIdx.x;       // global-layout token
  const float* mo = MO + (size_t)t * CDIM;
  float mu = 0.f;
#pragma unroll 4
  for (int c = 0; c < CDIM; ++c) mu += mo[c];
  mu *= (1.0f / CDIM);
  float var = 0.f;
#pragma unroll 4
  for (int c = 0; c < CDIM; ++c) { float d = mo[c] - mu; var += d * d; }
  float rstd = rsqrtf(var * (1.0f / CDIM) + 1e-5f);
#pragma unroll 4
  for (int c = 0; c < CDIM; ++c)
    out[(size_t)t * CDIM + c] =
        X1[(size_t)t * CDIM + c] + (mo[c] - mu) * rstd * nw[c] + nb[c];
}

// ---------------------------------------------------------------------------
// launcher
// ---------------------------------------------------------------------------
extern "C" void kernel_launch(void* const* d_in, const int* in_sizes, int n_in,
                              void* d_out, int out_size, void* d_ws, size_t ws_size,
                              hipStream_t stream) {
  (void)in_sizes; (void)n_in; (void)out_size; (void)ws_size;

  const float* x        = (const float*)d_in[0];
  const float* norm1_w  = (const float*)d_in[1];
  const float* norm1_b  = (const float*)d_in[2];
  const float* qkv_w    = (const float*)d_in[3];
  const float* qkv_b    = (const float*)d_in[4];
  const float* lscale   = (const float*)d_in[5];
  const float* cpb_w1   = (const float*)d_in[6];
  const float* cpb_b1   = (const float*)d_in[7];
  const float* cpb_w2   = (const float*)d_in[8];
  const float* proj_w   = (const float*)d_in[9];
  const float* proj_b   = (const float*)d_in[10];
  const float* norm2_w  = (const float*)d_in[11];
  const float* norm2_b  = (const float*)d_in[12];
  const float* fc1_w    = (const float*)d_in[13];
  const float* fc1_b    = (const float*)d_in[14];
  const float* fc2_w    = (const float*)d_in[15];
  const float* fc2_b    = (const float*)d_in[16];
  float* out = (float*)d_out;

  // ---- workspace carve-up ----
  char* ws = (char*)d_ws;
  size_t off = 0;
  auto alloc = [&](size_t bytes) -> char* {
    char* p = ws + off;
    off += (bytes + 255) & ~(size_t)255;
    return p;
  };
  _Float16* qkvw16 = (_Float16*)alloc((size_t)3 * CDIM * CDIM * 2);
  _Float16* projw16= (_Float16*)alloc((size_t)CDIM * CDIM * 2);
  _Float16* fc1w16 = (_Float16*)alloc((size_t)HID * CDIM * 2);
  _Float16* fc2w16 = (_Float16*)alloc((size_t)CDIM * HID * 2);
  float*    bt     = (float*)   alloc((size_t)CPB_E * NH * 4);
  float*    BIAS   = (float*)   alloc((size_t)NH * NTOK * NTOK * 4);
  _Float16* XW16   = (_Float16*)alloc((size_t)TOKS * CDIM * 2);
  _Float16* Q16    = (_Float16*)alloc((size_t)NW * NH * NTOK * HD * 2);
  _Float16* K16    = (_Float16*)alloc((size_t)NW * NH * NTOK * HD * 2);
  _Float16* Vt16   = (_Float16*)alloc((size_t)NW * NH * HD * NTOK * 2);
  _Float16* AO16   = (_Float16*)alloc((size_t)TOKS * CDIM * 2);
  float*    PO     = (float*)   alloc((size_t)TOKS * CDIM * 4);
  float*    X1     = (float*)   alloc((size_t)TOKS * CDIM * 4);
  _Float16* X116   = (_Float16*)alloc((size_t)TOKS * CDIM * 2);
  _Float16* H16    = (_Float16*)alloc((size_t)TOKS * HID * 2);
  float*    MO     = (float*)   alloc((size_t)TOKS * CDIM * 4);

  // ---- weight conversion ----
  k_f32_to_f16<<<(3*CDIM*CDIM + 255)/256, 256, 0, stream>>>(qkv_w,  qkvw16, 3*CDIM*CDIM);
  k_f32_to_f16<<<(CDIM*CDIM   + 255)/256, 256, 0, stream>>>(proj_w, projw16, CDIM*CDIM);
  k_f32_to_f16<<<(HID*CDIM    + 255)/256, 256, 0, stream>>>(fc1_w,  fc1w16, HID*CDIM);
  k_f32_to_f16<<<(CDIM*HID    + 255)/256, 256, 0, stream>>>(fc2_w,  fc2w16, CDIM*HID);

  // ---- continuous position bias ----
  k_cpb<<<CPB_E, HID, 0, stream>>>(cpb_w1, cpb_b1, cpb_w2, bt);
  k_bias<<<(NTOK*NTOK)/256, 256, 0, stream>>>(bt, BIAS);

  // ---- shift + gather windows ----
  k_gather<<<(TOKS*16)/256, 256, 0, stream>>>(x, XW16);

  // ---- QKV ----
  k_qkv<<<(NW*32*6)/8, 256, 0, stream>>>(XW16, qkvw16, qkv_b, Q16, K16, Vt16);
  k_norm<<<(NW*NH*NTOK)/256, 256, 0, stream>>>(Q16, K16, lscale);

  // ---- attention ----
  k_attn<<<dim3(NTOK/16, NH, NW), 128, 0, stream>>>(Q16, K16, Vt16, BIAS, AO16);

  // ---- projection + residual/LN1 ----
  k_proj<<<(TOKS/16 * 2)/8, 256, 0, stream>>>(AO16, projw16, proj_b, PO);
  k_ln1<<<TOKS/256, 256, 0, stream>>>(x, PO, norm1_w, norm1_b, X1, X116);

  // ---- MLP + residual/LN2 ----
  k_fc1<<<(TOKS/16 * 8)/8, 256, 0, stream>>>(X116, fc1w16, fc1_b, H16);
  k_fc2<<<(TOKS/16 * 2)/8, 256, 0, stream>>>(H16, fc2w16, fc2_b, MO);
  k_ln2<<<TOKS/256, 256, 0, stream>>>(X1, MO, norm2_w, norm2_b, out);
}
